// GraphUnet_68040871903660
// MI455X (gfx1250) — matrix operand order repaired and checked by reading, not compiled, not run on hardware
//
#include <hip/hip_runtime.h>
#include <hip/hip_bf16.h>
#include <math.h>

// ---------------------------------------------------------------------------
// Types for CDNA5 WMMA (wave32, 16x16x32 bf16 -> f32 accumulate)
// ---------------------------------------------------------------------------
typedef __attribute__((ext_vector_type(16))) __bf16 v16bf;
typedef __attribute__((ext_vector_type(8)))  __bf16 v8bf;
typedef __attribute__((ext_vector_type(8)))  float  v8f;

// ---------------------------------------------------------------------------
// Pack f32 matrix (rows x K) into bf16 (Rp x Kp), zero-padded. Kp,Rp mult 32.
// ---------------------------------------------------------------------------
__global__ void pack_bf16_kernel(const float* __restrict__ in, __bf16* __restrict__ out,
                                 int rows, int K, int Rp, int Kp)
{
    long t = blockIdx.x * (long)blockDim.x + threadIdx.x;
    long total = (long)Rp * Kp;
    if (t >= total) return;
    int k = (int)(t % Kp);
    int r = (int)(t / Kp);
    float v = (r < rows && k < K) ? in[(long)r * K + k] : 0.f;
    out[t] = (__bf16)v;
}

// ---------------------------------------------------------------------------
// m[c,n,f] = sum_mm g[n,mm] * h[c,mm,f]  written as bf16 with padded K (Kp),
// zero-filled pad. (n <= 32, tiny graph matmul; feeds the WMMA GEMM)
// ---------------------------------------------------------------------------
__global__ void gmul_bf16_kernel(const float* __restrict__ g, int Nn,
                                 const float* __restrict__ h, int C, int F, int Kp,
                                 __bf16* __restrict__ out)
{
    long t = blockIdx.x * (long)blockDim.x + threadIdx.x;
    long total = (long)C * Nn * Kp;
    if (t >= total) return;
    int f = (int)(t % Kp); long r = t / Kp;
    int n = (int)(r % Nn);
    int c = (int)(r / Nn);
    float acc = 0.f;
    if (f < F) {
        const float* grow = g + (long)n * Nn;
        const float* hc = h + (long)c * Nn * F + f;
        for (int m = 0; m < Nn; ++m) acc += grow[m] * hc[(long)m * F];
    }
    out[t] = (__bf16)acc;
}

// ---------------------------------------------------------------------------
// WMMA GEMM, bf16 inputs, f32 accumulate, fused bias + ReLU:
//   out[m][n] = relu( sum_k A[m*Kp+k] * W[n*Kp+k] + bias[n] )
// A: M x Kp bf16 (M multiple of 16, Kp multiple of 32, zero-padded K tail)
// W: Np x Kp bf16 (Np = ceil(N/32)*32, zero-padded rows/tail)
// One wave per (16 M) x (32 N): two 16x16 tiles sharing the A fragment.
// Guard-free inner loop: contiguous 16B v8bf loads in a contiguous-K order
// used by both A and B (A 16x32: lane m=L%16,g=L/16 holds K run kk+16g..+15;
// B 32x16: lane n=L%16 holds K run kk+16g..+15).
// ---------------------------------------------------------------------------
__global__ void wmma_gemm_bf16(const __bf16* __restrict__ A,
                               const __bf16* __restrict__ W,
                               const float* __restrict__ bias,
                               float* __restrict__ out,
                               int M, int N, int Kp)
{
    int lane = threadIdx.x & 31;
    int tg = blockIdx.x;              // group of two N tiles
    int tm = blockIdx.y;              // M tile
    int half = lane >> 4;
    int l16  = lane & 15;

    const __bf16* arp  = A + (long)(tm * 16 + l16) * Kp;
    const __bf16* brp0 = W + (long)(tg * 32 + l16) * Kp;
    const __bf16* brp1 = W + (long)(tg * 32 + 16 + l16) * Kp;

    v8f acc0 = {}, acc1 = {};
    for (int kk = 0; kk < Kp; kk += 32) {
        v8bf alo = *(const v8bf*)(arp + kk + 16 * half);
        v8bf ahi = *(const v8bf*)(arp + kk + 16 * half + 8);
        v16bf av = __builtin_shufflevector(alo, ahi,
            0, 1, 2, 3, 4, 5, 6, 7, 8, 9, 10, 11, 12, 13, 14, 15);

        v8bf b0l = *(const v8bf*)(brp0 + kk + 16 * half);
        v8bf b0h = *(const v8bf*)(brp0 + kk + 16 * half + 8);
        v16bf bv0 = __builtin_shufflevector(b0l, b0h,
            0, 1, 2, 3, 4, 5, 6, 7, 8, 9, 10, 11, 12, 13, 14, 15);

        v8bf b1l = *(const v8bf*)(brp1 + kk + 16 * half);
        v8bf b1h = *(const v8bf*)(brp1 + kk + 16 * half + 8);
        v16bf bv1 = __builtin_shufflevector(b1l, b1h,
            0, 1, 2, 3, 4, 5, 6, 7, 8, 9, 10, 11, 12, 13, 14, 15);

        acc0 = __builtin_amdgcn_wmma_f32_16x16x32_bf16(
            false, av, false, bv0, (short)0, acc0, false, false);
        acc1 = __builtin_amdgcn_wmma_f32_16x16x32_bf16(
            false, av, false, bv1, (short)0, acc1, false, false);
    }

    // Epilogue (C/D layout: lane col n=L%16; vgpr j -> row j + 8*(L/16))
    int bcol0 = tg * 32 + l16;
    int bcol1 = bcol0 + 16;
    if (bcol0 < N) {
        float bs = bias[bcol0];
        #pragma unroll
        for (int j = 0; j < 8; ++j) {
            int m = tm * 16 + j + 8 * half;
            out[(long)m * N + bcol0] = fmaxf(acc0[j] + bs, 0.f);
        }
    }
    if (bcol1 < N) {
        float bs = bias[bcol1];
        #pragma unroll
        for (int j = 0; j < 8; ++j) {
            int m = tm * 16 + j + 8 * half;
            out[(long)m * N + bcol1] = fmaxf(acc1[j] + bs, 0.f);
        }
    }
}

// ---------------------------------------------------------------------------
// Direct 3x3 conv + batchnorm + relu. Optional fused channel-concat input:
// channels [0,c1) come from in1, [c1,Cin) from in2. NCHW, pad=1.
// ---------------------------------------------------------------------------
__global__ void conv_bnrelu_kernel(const float* __restrict__ in1,
                                   const float* __restrict__ in2, int c1,
                                   int BB, int Cin, int H, int W,
                                   const float* __restrict__ wt,
                                   const float* __restrict__ bb,
                                   const float* __restrict__ gm,
                                   const float* __restrict__ bt,
                                   const float* __restrict__ rm,
                                   const float* __restrict__ rv,
                                   int Cout, float* __restrict__ out)
{
    long t = blockIdx.x * (long)blockDim.x + threadIdx.x;
    long total = (long)BB * Cout * H * W;
    if (t >= total) return;
    int x = (int)(t % W); long r = t / W;
    int y = (int)(r % H); r /= H;
    int co = (int)(r % Cout);
    int b  = (int)(r / Cout);
    int c2 = Cin - c1;

    float acc = 0.f;
    for (int ci = 0; ci < Cin; ++ci) {
        const float* plane = (ci < c1)
            ? in1 + ((long)b * c1 + ci) * H * W
            : in2 + ((long)b * c2 + (ci - c1)) * H * W;
        const float* wp = wt + ((long)co * Cin + ci) * 9;
        #pragma unroll
        for (int ky = 0; ky < 3; ++ky) {
            int yy = y + ky - 1;
            if (yy < 0 || yy >= H) continue;
            const float* rowp = plane + (long)yy * W;
            #pragma unroll
            for (int kx = 0; kx < 3; ++kx) {
                int xx = x + kx - 1;
                if (xx < 0 || xx >= W) continue;
                acc += rowp[xx] * wp[ky * 3 + kx];
            }
        }
    }
    float scale = gm[co] * rsqrtf(rv[co] + 1e-5f);
    float v = (acc + bb[co] - rm[co]) * scale + bt[co];
    out[t] = fmaxf(v, 0.f);
}

// ---------------------------------------------------------------------------
// maxpool over last dim, window 2, with argmax (0 = first wins ties, as jnp)
// ---------------------------------------------------------------------------
__global__ void maxpool_kernel(const float* __restrict__ in, float* __restrict__ out,
                               unsigned char* __restrict__ win, long total, int Fo)
{
    long t = blockIdx.x * (long)blockDim.x + threadIdx.x;
    if (t >= total) return;
    int j = (int)(t % Fo); long p = t / Fo;
    const float* ip = in + p * (long)(2 * Fo) + 2 * j;
    float v0 = ip[0], v1 = ip[1];
    out[t] = fmaxf(v0, v1);
    win[t] = (v1 > v0) ? 1 : 0;
}

// s[n] = sigmoid( b + mean_c( sum_f G[c,n,f]*w[f] ) )
__global__ void gpool_scores_kernel(const float* __restrict__ G,
                                    const float* __restrict__ w,
                                    const float* __restrict__ b,
                                    int C, int Nn, int F, float* __restrict__ s)
{
    int n = blockIdx.x;
    float p = 0.f;
    for (int i = threadIdx.x; i < C * F; i += blockDim.x) {
        int c = i / F, f = i - c * F;
        p += G[((long)c * Nn + n) * F + f] * w[f];
    }
    __shared__ float red[256];
    red[threadIdx.x] = p;
    __syncthreads();
    for (int st = 128; st; st >>= 1) {
        if ((int)threadIdx.x < st) red[threadIdx.x] += red[threadIdx.x + st];
        __syncthreads();
    }
    if (threadIdx.x == 0) {
        float v = b[0] + red[0] / (float)C;
        s[n] = 1.f / (1.f + expf(-v));
    }
}

// serial stable top-k (n <= 32), descending, ties -> lower index (lax.top_k)
__global__ void topk_kernel(const float* __restrict__ s, int n, int kk,
                            int* __restrict__ idx, float* __restrict__ vals)
{
    if (threadIdx.x != 0 || blockIdx.x != 0) return;
    bool used[64];
    for (int i = 0; i < n; ++i) used[i] = false;
    for (int j = 0; j < kk; ++j) {
        float best = -3.4e38f; int bi = 0;
        for (int i = 0; i < n; ++i) {
            float v = s[i];
            if (!used[i] && v > best) { best = v; bi = i; }
        }
        used[bi] = true; idx[j] = bi; vals[j] = best;
    }
}

// NH[c,j,f] = G[c, idx[j], f] * vals[j]
__global__ void gather_scale_kernel(const float* __restrict__ G,
                                    const int* __restrict__ idx,
                                    const float* __restrict__ vals,
                                    int C, int Nn, int kk, int F,
                                    float* __restrict__ out)
{
    long t = blockIdx.x * (long)blockDim.x + threadIdx.x;
    long total = (long)C * kk * F;
    if (t >= total) return;
    int f = (int)(t % F); long r = t / F;
    int j = (int)(r % kk);
    int c = (int)(r / kk);
    out[t] = G[((long)c * Nn + idx[j]) * F + f] * vals[j];
}

// ug2 = ((g!=0) @ (g!=0)) != 0
__global__ void ug2_kernel(const float* __restrict__ g, int n, float* __restrict__ out)
{
    int t = blockIdx.x * blockDim.x + threadIdx.x;
    if (t >= n * n) return;
    int i = t / n, j = t % n;
    float s = 0.f;
    for (int k = 0; k < n; ++k)
        s += (g[i * n + k] != 0.f && g[k * n + j] != 0.f) ? 1.f : 0.f;
    out[t] = (s != 0.f) ? 1.f : 0.f;
}

// gnew[i,j] = ug2[idx[i],idx[j]] / sum_i' ug2[idx[i'],idx[j]]   (kk<=16)
__global__ void pooledg_kernel(const float* __restrict__ ug2, int n,
                               const int* __restrict__ idx, int kk,
                               float* __restrict__ gnew)
{
    int t = threadIdx.x;
    if (t >= kk * kk) return;
    int i = t / kk, j = t % kk;
    float cs = 0.f;
    for (int ii = 0; ii < kk; ++ii) cs += ug2[idx[ii] * n + idx[j]];
    gnew[t] = ug2[idx[i] * n + idx[j]] / cs;
}

__global__ void invmap_kernel(const int* __restrict__ idx, int kk, int nfull,
                              int* __restrict__ inv)
{
    if (threadIdx.x == 0 && blockIdx.x == 0) {
        for (int i = 0; i < nfull; ++i) inv[i] = -1;
        for (int j = 0; j < kk; ++j) inv[idx[j]] = j;
    }
}

// out[c,nn,f] = pre[c,nn,f] + (inv[nn]>=0 ? h[c,inv[nn],f] : 0)
__global__ void gunpool_kernel(const float* __restrict__ hsmall,
                               const float* __restrict__ pre,
                               const int* __restrict__ inv,
                               int C, int nfull, int kk, int F,
                               float* __restrict__ out)
{
    long t = blockIdx.x * (long)blockDim.x + threadIdx.x;
    long total = (long)C * nfull * F;
    if (t >= total) return;
    int f = (int)(t % F); long r = t / F;
    int nn = (int)(r % nfull);
    int c  = (int)(r / nfull);
    float v = pre[t];
    int j = inv[nn];
    if (j >= 0) v += hsmall[((long)c * kk + j) * F + f];
    out[t] = v;
}

// uniform max-unpool (window 2): out[p, 2j+t] = (win[p,j]==t) ? in[p,j] : 0
__global__ void unpool_kernel(const float* __restrict__ in,
                              const unsigned char* __restrict__ win,
                              long planes, int F, float* __restrict__ out)
{
    long t = blockIdx.x * (long)blockDim.x + threadIdx.x;
    long total = planes * (long)(2 * F);
    if (t >= total) return;
    int x = (int)(t % (2 * F)); long p = t / (2 * F);
    int j = x >> 1, tt = x & 1;
    out[t] = ((int)win[p * F + j] == tt) ? in[p * F + j] : 0.f;
}

// per-batch unpool of a batch-uniform input using per-batch argmax indices
__global__ void unpool_batch_kernel(const float* __restrict__ in,   // (C,H,F)
                                    const unsigned char* __restrict__ win, // (B,C,H,F)
                                    int B, int C, int H, int F,
                                    float* __restrict__ out)        // (B,C,H,2F)
{
    long t = blockIdx.x * (long)blockDim.x + threadIdx.x;
    long total = (long)B * C * H * 2 * F;
    if (t >= total) return;
    int x = (int)(t % (2 * F)); long p = t / (2 * F); // p = (b*C+c)*H + y
    int y = (int)(p % H); long bc = p / H;
    int c = (int)(bc % C);
    int j = x >> 1, tt = x & 1;
    float v = ((int)win[p * F + j] == tt) ? in[((long)c * H + y) * F + j] : 0.f;
    out[t] = v;
}

// ---------------------------------------------------------------------------
// Host orchestration
// ---------------------------------------------------------------------------
struct CnnP { const float *w, *b, *gamma, *beta, *rm, *rv; };
struct GcnP { const float *w1, *b1, *w2, *b2; };
struct GpP  { const float *w, *b; };

extern "C" void kernel_launch(void* const* d_in, const int* in_sizes, int n_in,
                              void* d_out, int out_size, void* d_ws, size_t ws_size,
                              hipStream_t stream)
{
    const float* g0 = (const float*)d_in[0];   // (16,32,32); only batch 0 matters
    const float* h0 = (const float*)d_in[1];   // (16,32,3000)
    auto F = [&](int i) { return (const float*)d_in[i]; };

    CnnP dcnn[4], ucnn[6], bot1, bot2, lastc;
    GcnP dgcn[2], ugcn[2], bgcn;
    GpP  gp[2];

    bool sortedOrder = (n_in > 2 && in_sizes[2] == 32); // sorted: bottom1.b (32); insertion: down_cnn[0].w (72)
    if (sortedOrder) {
        auto cnn = [&](int base) { CnnP c{F(base + 5), F(base), F(base + 2), F(base + 1), F(base + 3), F(base + 4)}; return c; };
        bot1 = cnn(2); bot2 = cnn(8);
        bgcn = {F(16), F(14), F(17), F(15)};
        dcnn[0] = cnn(18); dcnn[1] = cnn(24); dcnn[2] = cnn(30); dcnn[3] = cnn(36);
        dgcn[0] = {F(44), F(42), F(45), F(43)};
        dgcn[1] = {F(48), F(46), F(49), F(47)};
        gp[0] = {F(51), F(50)}; gp[1] = {F(53), F(52)};
        lastc = cnn(54);
        for (int i = 0; i < 6; ++i) ucnn[i] = cnn(60 + 6 * i);
        ugcn[0] = {F(98), F(96), F(99), F(97)};
        ugcn[1] = {F(102), F(100), F(103), F(101)};
    } else {
        auto cnn = [&](int base) { CnnP c{F(base), F(base + 1), F(base + 2), F(base + 3), F(base + 4), F(base + 5)}; return c; };
        dcnn[0] = cnn(2); dcnn[1] = cnn(8); dcnn[2] = cnn(14); dcnn[3] = cnn(20);
        dgcn[0] = {F(26), F(27), F(28), F(29)};
        dgcn[1] = {F(30), F(31), F(32), F(33)};
        gp[0] = {F(34), F(35)}; gp[1] = {F(36), F(37)};
        for (int i = 0; i < 6; ++i) ucnn[i] = cnn(38 + 6 * i);
        ugcn[0] = {F(74), F(75), F(76), F(77)};
        ugcn[1] = {F(78), F(79), F(80), F(81)};
        bot1 = cnn(82); bot2 = cnn(88);
        bgcn = {F(94), F(95), F(96), F(97)};
        lastc = cnn(98);
    }

    // ---- workspace layout (bump allocator, floats, 64B aligned chunks) ----
    float* ws = (float*)d_ws;
    size_t off = 0;
    auto alloc = [&](size_t n) { float* p = ws + off; off += (n + 15) & ~((size_t)15); return p; };
    float* big1 = alloc(12288000);   // (16,8,32,3000): conv1 out, later unpool out, later V2
    float* C0   = alloc(12288000);   // couts[0], kept until final concat-conv
    float* big2 = alloc(12288000);   // V1 (post-concat conv)
    float* P0   = alloc(6144000);    // (16,8,32,1500)
    unsigned char* W0 = (unsigned char*)alloc(1536000); // 6,144,000 bytes argmax
    __bf16* Abf  = (__bf16*)alloc(192512);  // 385,024 bf16: A (M=256 x Kp<=1504)
    __bf16* Wbf1 = (__bf16*)alloc(192512);  // w1 packed (256 x Kp<=1504)
    __bf16* Wbf2 = (__bf16*)alloc(192512);  // w2 packed (Np<=1504 x 256)
    float* TMPB = alloc(384000);     // h1 f32 (C*Nn x 256)
    float* G0   = alloc(384000);     // gouts[0] (8,32,1500) uniform
    float* NH0  = alloc(192000);     // (8,16,1500)
    float* T1a  = alloc(384000);
    float* C1   = alloc(384000);     // couts[1] (16,16,1500) uniform
    float* P1   = alloc(192000);
    unsigned char* W1 = (unsigned char*)alloc(48000);   // 192,000 bytes
    float* G1   = alloc(192000);     // gouts[1] (16,16,750)
    float* NH1  = alloc(96000);      // (16,8,750)
    float* Bt1  = alloc(192000);
    float* Bt2  = alloc(192000);
    float* GB   = alloc(192000);     // bottom gcn out (32,8,750)
    float* U1a  = alloc(96000);
    float* U1b  = alloc(192000);
    float* U1c  = alloc(192000);
    float* U1d  = alloc(384000);
    float* U1e  = alloc(384000);
    float* U1f  = alloc(384000);
    float* U0a  = alloc(192000);
    float* U0b  = alloc(384000);
    float* U0c  = alloc(384000);     // uniform pre-unpool (8,32,1500)
    float* g1   = alloc(256);
    float* g2   = alloc(64);
    float* ug2b = alloc(1024);
    float* sc   = alloc(64);
    float* vals0 = alloc(16);
    float* vals1 = alloc(16);
    int* idx0 = (int*)alloc(16);
    int* idx1 = (int*)alloc(16);
    int* inv  = (int*)alloc(64);
    (void)ws_size; (void)out_size;

    auto cdiv = [](long a, long b) { return (unsigned)((a + b - 1) / b); };
    auto al32 = [](int x) { return (x + 31) & ~31; };
    auto conv = [&](const float* i1, const float* i2, int c1, int BB, int Cin,
                    int H, int W, const CnnP& p, int Cout, float* o) {
        long total = (long)BB * Cout * H * W;
        conv_bnrelu_kernel<<<dim3(cdiv(total, 256)), dim3(256), 0, stream>>>(
            i1, i2, c1, BB, Cin, H, W, p.w, p.b, p.gamma, p.beta, p.rm, p.rv, Cout, o);
    };
    auto pack = [&](const float* in, __bf16* o, int rows, int K, int Rp, int Kp) {
        long total = (long)Rp * Kp;
        pack_bf16_kernel<<<dim3(cdiv(total, 256)), dim3(256), 0, stream>>>(in, o, rows, K, Rp, Kp);
    };
    auto gemm = [&](const __bf16* A, const __bf16* W, const float* bias,
                    float* o, int M, int N, int Kp) {
        wmma_gemm_bf16<<<dim3(cdiv(N, 32), (unsigned)(M / 16)), dim3(32), 0, stream>>>(
            A, W, bias, o, M, N, Kp);
    };
    auto gmulb = [&](const float* gpt, int Nn, const float* hin, int C, int Fd, int Kp, __bf16* o) {
        long total = (long)C * Nn * Kp;
        gmul_bf16_kernel<<<dim3(cdiv(total, 256)), dim3(256), 0, stream>>>(gpt, Nn, hin, C, Fd, Kp, o);
    };
    // gcn(): m = g @ h; h1 = relu(m @ w1^T + b1); m2 = g @ h1; out = relu(m2 @ w2^T + b2)
    auto gcnrun = [&](const float* gpt, int Nn, int C, int Fd, const GcnP& p,
                      const float* hin, float* o) {
        int M = C * Nn;
        int Kp1 = al32(Fd);
        gmulb(gpt, Nn, hin, C, Fd, Kp1, Abf);           // m -> bf16, K-padded
        pack(p.w1, Wbf1, 256, Fd, 256, Kp1);            // w1 (256 x Fd) -> bf16
        gemm(Abf, Wbf1, p.b1, TMPB, M, 256, Kp1);       // h1 = relu(m @ w1^T + b1)
        gmulb(gpt, Nn, TMPB, C, 256, 256, Abf);         // m2 -> bf16
        int Np2 = al32(Fd);
        pack(p.w2, Wbf2, Fd, 256, Np2, 256);            // w2 (Fd x 256) -> bf16, row-padded
        gemm(Abf, Wbf2, p.b2, o, M, Fd, 256);           // out = relu(m2 @ w2^T + b2)
    };

    // ================= Down level 0 (per batch) =================
    conv(h0, nullptr, 1, 16, 1, 32, 3000, dcnn[0], 8, big1);
    conv(big1, nullptr, 8, 16, 8, 32, 3000, dcnn[1], 8, C0);       // couts[0]
    {
        long total = 16L * 8 * 32 * 1500;
        maxpool_kernel<<<dim3(cdiv(total, 256)), dim3(256), 0, stream>>>(C0, P0, W0, total, 1500);
    }
    // ================= Uniform middle (batch-0 math, broadcast-exact) ======
    gcnrun(g0, 32, 8, 1500, dgcn[0], P0, G0);                       // gouts[0]
    gpool_scores_kernel<<<dim3(32), dim3(256), 0, stream>>>(G0, gp[0].w, gp[0].b, 8, 32, 1500, sc);
    topk_kernel<<<1, 1, 0, stream>>>(sc, 32, 16, idx0, vals0);
    {
        long total = 8L * 16 * 1500;
        gather_scale_kernel<<<dim3(cdiv(total, 256)), dim3(256), 0, stream>>>(G0, idx0, vals0, 8, 32, 16, 1500, NH0);
    }
    ug2_kernel<<<dim3(4), dim3(256), 0, stream>>>(g0, 32, ug2b);
    pooledg_kernel<<<dim3(1), dim3(256), 0, stream>>>(ug2b, 32, idx0, 16, g1);

    // ---- Down level 1 ----
    conv(NH0, nullptr, 8, 1, 8, 16, 1500, dcnn[2], 16, T1a);
    conv(T1a, nullptr, 16, 1, 16, 16, 1500, dcnn[3], 16, C1);      // couts[1]
    {
        long total = 16L * 16 * 750;
        maxpool_kernel<<<dim3(cdiv(total, 256)), dim3(256), 0, stream>>>(C1, P1, W1, total, 750);
    }
    gcnrun(g1, 16, 16, 750, dgcn[1], P1, G1);                       // gouts[1]
    gpool_scores_kernel<<<dim3(16), dim3(256), 0, stream>>>(G1, gp[1].w, gp[1].b, 16, 16, 750, sc);
    topk_kernel<<<1, 1, 0, stream>>>(sc, 16, 8, idx1, vals1);
    {
        long total = 16L * 8 * 750;
        gather_scale_kernel<<<dim3(cdiv(total, 256)), dim3(256), 0, stream>>>(G1, idx1, vals1, 16, 16, 8, 750, NH1);
    }
    ug2_kernel<<<dim3(1), dim3(256), 0, stream>>>(g1, 16, ug2b);
    pooledg_kernel<<<dim3(1), dim3(256), 0, stream>>>(ug2b, 16, idx1, 8, g2);

    // ---- Bottom ----
    conv(NH1, nullptr, 16, 1, 16, 8, 750, bot1, 32, Bt1);
    conv(Bt1, nullptr, 32, 1, 32, 8, 750, bot2, 32, Bt2);
    gcnrun(g2, 8, 32, 750, bgcn, Bt2, GB);

    // ---- Up u=1 ----
    conv(GB, nullptr, 32, 1, 32, 8, 750, ucnn[3], 16, U1a);
    invmap_kernel<<<1, 1, 0, stream>>>(idx1, 8, 16, inv);
    {
        long total = 16L * 16 * 750;
        gunpool_kernel<<<dim3(cdiv(total, 256)), dim3(256), 0, stream>>>(U1a, G1, inv, 16, 16, 8, 750, U1b);
    }
    gcnrun(g1, 16, 16, 750, ugcn[1], U1b, U1c);
    {
        long total = 16L * 16 * 1500;
        unpool_kernel<<<dim3(cdiv(total, 256)), dim3(256), 0, stream>>>(U1c, W1, 16L * 16, 750, U1d);
    }
    conv(U1d, C1, 16, 1, 32, 16, 1500, ucnn[4], 16, U1e);          // fused concat
    conv(U1e, nullptr, 16, 1, 16, 16, 1500, ucnn[5], 16, U1f);

    // ---- Up u=0 (uniform part) ----
    conv(U1f, nullptr, 16, 1, 16, 16, 1500, ucnn[0], 8, U0a);
    invmap_kernel<<<1, 1, 0, stream>>>(idx0, 16, 32, inv);
    {
        long total = 8L * 32 * 1500;
        gunpool_kernel<<<dim3(cdiv(total, 256)), dim3(256), 0, stream>>>(U0a, G0, inv, 8, 32, 16, 1500, U0b);
    }
    gcnrun(g0, 32, 8, 1500, ugcn[0], U0b, U0c);

    // ================= Final per-batch stage =================
    {
        long total = 16L * 8 * 32 * 3000;
        unpool_batch_kernel<<<dim3(cdiv(total, 256)), dim3(256), 0, stream>>>(U0c, W0, 16, 8, 32, 1500, big1);
    }
    conv(big1, C0, 8, 16, 16, 32, 3000, ucnn[1], 8, big2);         // fused concat with couts[0]
    conv(big2, nullptr, 8, 16, 8, 32, 3000, ucnn[2], 8, big1);
    conv(big1, nullptr, 8, 16, 8, 32, 3000, lastc, 1, (float*)d_out);
}
